// ViewLearner_88064009437326
// MI455X (gfx1250) — compile-verified, matrix operand order checked by reading, not run on hardware
//
#include <hip/hip_runtime.h>
#include <hip/hip_bf16.h>

#define N_NODES 50000
#define N_EDGES 800000
#define D 64
#define MLP 64

typedef __attribute__((ext_vector_type(16))) __bf16 v16bf;
typedef __attribute__((ext_vector_type(8)))  __bf16 v8bf;
typedef __attribute__((ext_vector_type(8)))  float  v8f;
typedef __attribute__((ext_vector_type(4)))  float  v4f;

// Fragment element i (0..15) of a 32-wide K chunk maps to K = 8*hi + (i<8 ? i : i+8).
// (Two contiguous 8-element runs: [8hi, 8hi+8) and [16+8hi, 16+8hi+8).)
__device__ __forceinline__ int koff(int i, int hi) {
  return 8 * hi + (i < 8 ? i : i + 8);
}

__global__ void k_zero(float* __restrict__ p, int n) {
  int i = blockIdx.x * blockDim.x + threadIdx.x;
  if (i < n) p[i] = 0.0f;
}

// agg[dst] += x[src] over all edges. 16 lanes per edge, 4 floats per lane.
__global__ void k_scatter(const float* __restrict__ x, const int* __restrict__ ei,
                          float* __restrict__ agg) {
  int t = blockIdx.x * blockDim.x + threadIdx.x;
  if (t >= N_EDGES * 16) return;
  int e  = t >> 4;
  int d4 = (t & 15) * 4;
  int s  = ei[e];
  int dd = ei[N_EDGES + e];
  const float4 v = *(const float4*)(x + (size_t)s * D + d4);
  float* q = agg + (size_t)dd * D + d4;
  unsafeAtomicAdd(q + 0, v.x);
  unsafeAtomicAdd(q + 1, v.y);
  unsafeAtomicAdd(q + 2, v.z);
  unsafeAtomicAdd(q + 3, v.w);
}

// node_emb = relu((x + agg) @ W_enc + b_enc), stored bf16.
// 128 threads = 4 waves, each wave a 16-node tile, 4 N-tiles of 16, K=64 (2 chunks).
__global__ void __launch_bounds__(128)
k_node(const float* __restrict__ x, const float* __restrict__ agg,
       const float* __restrict__ Wenc, const float* __restrict__ benc,
       __bf16* __restrict__ nemb) {
  // W_enc pre-swizzled into per-lane WMMA B fragments: [t][c][lane][16]
  __shared__ alignas(32) __bf16 Wf[4 * 2 * 32 * 16];   // 4096 = D*D, 8 KB
  __shared__ float bs[D];
  const int tid = threadIdx.x;
  for (int f = tid; f < 4 * 2 * 32; f += 128) {        // 256 fragments
    int lane_ = f & 31, c = (f >> 5) & 1, t = f >> 6;
    int hi_ = lane_ >> 4, n = 16 * t + (lane_ & 15);
    v16bf tmp;
#pragma unroll
    for (int i = 0; i < 16; ++i)
      tmp[i] = (__bf16)Wenc[(32 * c + koff(i, hi_)) * D + n];
    *(v16bf*)&Wf[f * 16] = tmp;
  }
  if (tid < D) bs[tid] = benc[tid];
  __syncthreads();

  const int lane = tid & 31, wave = tid >> 5;
  const int hi = lane >> 4, ml = lane & 15;
  int tile = blockIdx.x * 4 + wave;
  const int maxTile = N_NODES / 16 - 1;                // 3125 tiles exact
  if (tile > maxTile) tile = maxTile;                  // clamp: EXEC stays all-ones
  const int tb  = tile * 16;
  const int row = tb + ml;

  const float* xr = x   + (size_t)row * D;
  const float* ar = agg + (size_t)row * D;
  v16bf afrag[2];
#pragma unroll
  for (int c = 0; c < 2; ++c) {
#pragma unroll
    for (int r = 0; r < 2; ++r) {                      // two contiguous 8-elt runs
      int k0 = 32 * c + 16 * r + 8 * hi;
      v4f x0 = *(const v4f*)(xr + k0);
      v4f x1 = *(const v4f*)(xr + k0 + 4);
      v4f a0 = *(const v4f*)(ar + k0);
      v4f a1 = *(const v4f*)(ar + k0 + 4);
#pragma unroll
      for (int i = 0; i < 4; ++i) {
        afrag[c][8 * r + i]     = (__bf16)(x0[i] + a0[i]);
        afrag[c][8 * r + 4 + i] = (__bf16)(x1[i] + a1[i]);
      }
    }
  }

#pragma unroll
  for (int t = 0; t < 4; ++t) {
    v8f acc = {};
#pragma unroll
    for (int c = 0; c < 2; ++c) {
      v16bf b = *(const v16bf*)&Wf[((t * 2 + c) * 32 + lane) * 16];
      acc = __builtin_amdgcn_wmma_f32_16x16x32_bf16(
          false, afrag[c], false, b, (short)0, acc, false, false);
    }
    const int n = 16 * t + ml;
#pragma unroll
    for (int v = 0; v < 8; ++v) {
      int m = v + (hi ? 8 : 0);
      float r = acc[v] + bs[n];
      r = r > 0.0f ? r : 0.0f;
      nemb[(size_t)(tb + m) * D + n] = (__bf16)r;
    }
  }
}

// Per edge: h = relu(concat(nemb[src], nemb[dst]) @ W1 + b1); out = h @ W2 + b2.
// 128 threads = 4 waves, each wave 16 edges (K=128 bf16 -> 4 WMMA K-chunks).
// Final 64-wide dot done in-register with shfl_xor reduction (no LDS staging).
__global__ void __launch_bounds__(128)
k_edge(const __bf16* __restrict__ nemb, const int* __restrict__ ei,
       const float* __restrict__ W1, const float* __restrict__ b1,
       const float* __restrict__ W2, const float* __restrict__ b2,
       float* __restrict__ out) {
  // W1 pre-swizzled into per-lane WMMA B fragments: [t][c][lane][16]
  __shared__ alignas(32) __bf16 W1f[4 * 4 * 32 * 16]; // 8192 = 2*D*MLP, 16 KB
  __shared__ float b1s[MLP];
  __shared__ float W2s[MLP];

  const int tid = threadIdx.x;
  for (int f = tid; f < 4 * 4 * 32; f += 128) {       // 512 fragments
    int lane_ = f & 31, c = (f >> 5) & 3, t = f >> 7;
    int hi_ = lane_ >> 4, n = 16 * t + (lane_ & 15);
    v16bf tmp;
#pragma unroll
    for (int i = 0; i < 16; ++i)
      tmp[i] = (__bf16)W1[(32 * c + koff(i, hi_)) * MLP + n];
    *(v16bf*)&W1f[f * 16] = tmp;
  }
  if (tid < MLP) { b1s[tid] = b1[tid]; W2s[tid] = W2[tid]; }
  __syncthreads();

  const int lane = tid & 31, wave = tid >> 5;
  const int hi = lane >> 4, ml = lane & 15;
  const long long eb = ((long long)blockIdx.x * 4 + wave) * 16;
  long long e = eb + ml;
  if (e > (long long)N_EDGES - 1) e = (long long)N_EDGES - 1;
  const int s  = ei[e];
  const int dn = ei[N_EDGES + e];
  const __bf16* se = nemb + (size_t)s  * D;
  const __bf16* de = nemb + (size_t)dn * D;

  // A fragments: per chunk, two aligned 16-byte runs from the gathered row.
  v16bf afrag[4];
#pragma unroll
  for (int c = 0; c < 4; ++c) {
    const __bf16* src = (c < 2) ? se : de;
    int base = 32 * (c & 1) + 8 * hi;
    v8bf r0 = *(const v8bf*)(src + base);
    v8bf r1 = *(const v8bf*)(src + base + 16);
#pragma unroll
    for (int i = 0; i < 8; ++i) { afrag[c][i] = r0[i]; afrag[c][i + 8] = r1[i]; }
  }

  float part[8];
#pragma unroll
  for (int v = 0; v < 8; ++v) part[v] = 0.0f;

#pragma unroll
  for (int t = 0; t < 4; ++t) {
    v8f acc = {};
#pragma unroll
    for (int c = 0; c < 4; ++c) {
      v16bf b = *(const v16bf*)&W1f[((t * 4 + c) * 32 + lane) * 16];
      acc = __builtin_amdgcn_wmma_f32_16x16x32_bf16(
          false, afrag[c], false, b, (short)0, acc, false, false);
    }
    const int n = 16 * t + ml;
    const float w2 = W2s[n], bb = b1s[n];
#pragma unroll
    for (int v = 0; v < 8; ++v) {
      float r = acc[v] + bb;
      part[v] += (r > 0.0f ? r : 0.0f) * w2;
    }
  }

  // Row m = v + 8*hi lives entirely in one 16-lane half: xor-reduce within it.
#pragma unroll
  for (int v = 0; v < 8; ++v) {
    float sv = part[v];
    sv += __shfl_xor(sv, 1);
    sv += __shfl_xor(sv, 2);
    sv += __shfl_xor(sv, 4);
    sv += __shfl_xor(sv, 8);
    part[v] = sv;
  }

  const float bias2 = b2[0];
  if (ml == 0) {                                      // lanes 0 (m=0..7) and 16 (m=8..15)
    long long base = eb + (hi ? 8 : 0);
    if (base + 7 < (long long)N_EDGES) {
      float4 o0 = make_float4(part[0] + bias2, part[1] + bias2,
                              part[2] + bias2, part[3] + bias2);
      float4 o1 = make_float4(part[4] + bias2, part[5] + bias2,
                              part[6] + bias2, part[7] + bias2);
      *(float4*)(out + base)     = o0;
      *(float4*)(out + base + 4) = o1;
    }
  }
}

extern "C" void kernel_launch(void* const* d_in, const int* in_sizes, int n_in,
                              void* d_out, int out_size, void* d_ws, size_t ws_size,
                              hipStream_t stream) {
  const float* x    = (const float*)d_in[0];
  const int*   ei   = (const int*)  d_in[1];
  const float* Wenc = (const float*)d_in[2];
  const float* benc = (const float*)d_in[3];
  const float* W1   = (const float*)d_in[4];
  const float* b1   = (const float*)d_in[5];
  const float* W2   = (const float*)d_in[6];
  const float* b2   = (const float*)d_in[7];
  float* out = (float*)d_out;

  // workspace: agg f32 [N*D] then node_emb bf16 [N*D]
  float* agg = (float*)d_ws;
  size_t aggBytes = ((size_t)N_NODES * D * sizeof(float) + 255) & ~(size_t)255;
  __bf16* nemb = (__bf16*)((char*)d_ws + aggBytes);

  {
    int n = N_NODES * D;
    k_zero<<<(n + 255) / 256, 256, 0, stream>>>(agg, n);
  }
  {
    int threads = N_EDGES * 16;
    k_scatter<<<(threads + 255) / 256, 256, 0, stream>>>(x, ei, agg);
  }
  {
    int tiles = N_NODES / 16;                 // 3125
    int blocks = (tiles + 3) / 4;             // 782
    k_node<<<blocks, 128, 0, stream>>>(x, agg, Wenc, benc, nemb);
  }
  {
    int blocks = (N_EDGES + 63) / 64;         // 12500
    k_edge<<<blocks, 128, 0, stream>>>(nemb, ei, W1, b1, W2, b2, out);
  }
}
